// EAGCN_16200616640806
// MI455X (gfx1250) — matrix operand order repaired and checked by reading, not compiled, not run on hardware
//
#include <hip/hip_runtime.h>
#include <hip/hip_bf16.h>
#include <stdint.h>

// ---------------- CDNA5 WMMA types ----------------
typedef __attribute__((ext_vector_type(16))) __bf16 v16bf;
typedef __attribute__((ext_vector_type(8)))  float  v8f;

#define Nn 32
#define Cc 256
#define HW 1024
#define Ss 64

__device__ __forceinline__ uint16_t f2bf(float x) {
  uint32_t u = __float_as_uint(x);
  uint32_t r = u + 0x7FFFu + ((u >> 16) & 1u);   // round-to-nearest-even
  return (uint16_t)(r >> 16);
}

union FragAB { v16bf v; uint32_t u[8]; };

// NT GEMM tile: A row-major [M][K], Bt row-major [N][K] (both bf16, K contiguous).
// One wave computes one 16x16 f32 tile, accumulating over K in steps of 32 with
// v_wmma_f32_16x16x32_bf16. Fragment lane layouts per CDNA5 ISA 7.12.2.
__device__ __forceinline__ v8f wmma_nt_tile(const uint16_t* __restrict__ A,
                                            const uint16_t* __restrict__ Bt,
                                            int lda, int ldb, int m0, int n0, int K) {
  const int lane = threadIdx.x & 31;
  const int half = lane >> 4;     // lane group 0/1
  const int idx  = lane & 15;     // M index for A, N index for B
  const uint16_t* arow = A  + (size_t)(m0 + idx) * lda;
  const uint16_t* brow = Bt + (size_t)(n0 + idx) * ldb;
  v8f acc = {};
  for (int k0 = 0; k0 < K; k0 += 32) {
    FragAB fa, fb;
#pragma unroll
    for (int j = 0; j < 8; ++j) {
      const int ka = k0 + ((j & 3) << 1) + ((j >> 2) << 4) + (half << 3);
      const int kb = k0 + ((j & 3) << 1) + ((j >> 2) << 4) + (half << 3);
      fa.u[j] = *reinterpret_cast<const uint32_t*>(arow + ka);
      fb.u[j] = *reinterpret_cast<const uint32_t*>(brow + kb);
    }
    acc = __builtin_amdgcn_wmma_f32_16x16x32_bf16(false, fa.v, false, fb.v,
                                                  (short)0, acc, false, false);
  }
  return acc;
}

// ---------------- prep kernels ----------------
__global__ void k_zero(float* p, int count) {
  int i = blockIdx.x * blockDim.x + threadIdx.x;
  if (i < count) p[i] = 0.f;
}

__global__ void k_f2bf(const float* __restrict__ src, uint16_t* __restrict__ dst, int count) {
  int i = blockIdx.x * blockDim.x + threadIdx.x;
  if (i < count) dst[i] = f2bf(src[i]);
}

// seg (n,c,hw) fp32 -> seg_bf (n,c,hw) bf16 + segT_bf (n,hw,c) bf16 + per-(n,c) row sums
__global__ void k_trans(const float* __restrict__ seg, uint16_t* __restrict__ seg_bf,
                        uint16_t* __restrict__ segT_bf, float* __restrict__ rowsum) {
  __shared__ float tile[32][33];
  __shared__ float part[32][8];
  const int tx = threadIdx.x, ty = threadIdx.y;
  const int p0 = blockIdx.x * 32, c0 = blockIdx.y * 32, n = blockIdx.z;
#pragma unroll
  for (int i = 0; i < 4; ++i) {
    const int r = ty + i * 8;
    const size_t gi = ((size_t)n * Cc + (c0 + r)) * HW + p0 + tx;
    const float v = seg[gi];
    tile[r][tx] = v;
    seg_bf[gi] = f2bf(v);
  }
  __syncthreads();
  float s = 0.f;
#pragma unroll
  for (int k = 0; k < 4; ++k) s += tile[tx][ty * 4 + k];
  part[tx][ty] = s;
#pragma unroll
  for (int i = 0; i < 4; ++i) {
    const int r = ty + i * 8;
    segT_bf[((size_t)n * HW + (p0 + r)) * Cc + c0 + tx] = f2bf(tile[tx][r]);
  }
  __syncthreads();
  if (ty == 0) {
    float t = 0.f;
#pragma unroll
    for (int k = 0; k < 8; ++k) t += part[tx][k];
    atomicAdd(rowsum + n * Cc + c0 + tx, t);
  }
}

// column max-pools over c: seg_ss[q] = max_c(seg)*w_s2+b_s2 ; edge_mm[p] = max_c(seg*edge)*w_s3+b_s3
__global__ void k_prep_cols(const float* __restrict__ seg, const float* __restrict__ edge,
                            const float* __restrict__ w_s2, const float* __restrict__ b_s2,
                            const float* __restrict__ w_s3, const float* __restrict__ b_s3,
                            float* __restrict__ seg_ss, float* __restrict__ edge_mm) {
  const int n = blockIdx.y;
  const int p = blockIdx.x * blockDim.x + threadIdx.x;
  const float* segn = seg  + (size_t)n * Cc * HW;
  const float* edgn = edge + (size_t)n * Cc * HW;
  float ms = -3.4e38f, me = -3.4e38f;
  for (int c = 0; c < Cc; ++c) {
    const float sv = segn[(size_t)c * HW + p];
    const float ev = edgn[(size_t)c * HW + p];
    ms = fmaxf(ms, sv);
    me = fmaxf(me, sv * ev);
  }
  seg_ss[n * HW + p]  = ms * w_s2[0] + b_s2[0];
  edge_mm[n * HW + p] = me * w_s3[0] + b_s3[0];
}

// ch_att = relu(mean_hw(seg) @ W_mlp^T + b_mlp)
__global__ void k_chatt(const float* __restrict__ rowsum, const float* __restrict__ Wmlp,
                        const float* __restrict__ bmlp, float* __restrict__ chatt) {
  const int n = blockIdx.x, s = threadIdx.x;
  const float* rs = rowsum + n * Cc;
  const float* w  = Wmlp + s * Cc;
  float acc = bmlp[s];
  for (int c = 0; c < Cc; ++c) acc += (rs[c] * (1.0f / HW)) * w[c];
  chatt[n * Ss + s] = fmaxf(acc, 0.f);
}

// ---------------- GEMM kernels (bf16 WMMA, f32 accumulate) ----------------
// W(64x256) @ seg(256x1024): A=W_bf, Bt=segT_bf. mode0 -> fp32 seg_s; mode1 -> bf16 seg_c + seg_c^T
__global__ void __launch_bounds__(256)
k_gemm_sc(const uint16_t* __restrict__ A, const uint16_t* __restrict__ segT,
          const float* __restrict__ bias, float* __restrict__ outF,
          uint16_t* __restrict__ outB, uint16_t* __restrict__ outBT, int mode) {
  const int n = blockIdx.y;
  const int tile = blockIdx.x * 8 + (threadIdx.x >> 5);
  const int m0 = (tile >> 6) * 16;
  const int n0 = (tile & 63) * 16;
  const uint16_t* Bt = segT + (size_t)n * HW * Cc;
  v8f acc = wmma_nt_tile(A, Bt, Cc, Cc, m0, n0, Cc);
  const int lane = threadIdx.x & 31, half = lane >> 4, col = lane & 15;
#pragma unroll
  for (int r = 0; r < 8; ++r) {
    const int m = m0 + r + half * 8;
    const int q = n0 + col;
    const float v = acc[r] + bias[m];
    if (mode == 0) {
      outF[((size_t)n * Ss + m) * HW + q] = v;
    } else {
      const uint16_t bv = f2bf(v);
      outB [((size_t)n * Ss + m) * HW + q] = bv;
      outBT[((size_t)n * HW + q) * Ss + m] = bv;
    }
  }
}

// u[p] = sum_t seg_s_flat[p*64+t]*a[t]; v[q] = sum_t a[t]*seg_s[t][q]; plus max/min(u)
__global__ void __launch_bounds__(1024)
k_uv(const float* __restrict__ segs, const float* __restrict__ chatt,
     float* __restrict__ uvec, float* __restrict__ vvec, float* __restrict__ umm) {
  __shared__ float sA[Ss];
  __shared__ float red[1024];
  const int n = blockIdx.x, t = threadIdx.x;
  if (t < Ss) sA[t] = chatt[n * Ss + t];
  __syncthreads();
  const float* flat = segs + (size_t)n * Ss * HW;
  float u = 0.f;
  const float4* fp = reinterpret_cast<const float4*>(flat + (size_t)t * Ss);
#pragma unroll
  for (int j = 0; j < 16; ++j) {
    const float4 x = fp[j];
    u += x.x * sA[4*j] + x.y * sA[4*j+1] + x.z * sA[4*j+2] + x.w * sA[4*j+3];
  }
  uvec[n * HW + t] = u;
  float v = 0.f;
  for (int j = 0; j < Ss; ++j) v += sA[j] * flat[(size_t)j * HW + t];
  vvec[n * HW + t] = v;
  red[t] = u; __syncthreads();
  for (int s = 512; s > 0; s >>= 1) { if (t < s) red[t] = fmaxf(red[t], red[t+s]); __syncthreads(); }
  const float umax = red[0]; __syncthreads();
  red[t] = u; __syncthreads();
  for (int s = 512; s > 0; s >>= 1) { if (t < s) red[t] = fminf(red[t], red[t+s]); __syncthreads(); }
  if (t == 0) { umm[n*2] = umax; umm[n*2+1] = red[0]; }
}

// ---------------- fused sigma GEMM + dual softmax ----------------
// One block owns q-strip [q0, q0+16) x all 1024 p. 8 waves x 8 p-tiles each:
// sigma_out^T[q][p] = sum_t seg_c^T[q][t] * seg_c_flat[p*64+t]  (K=64, WMMA)
// stays entirely in registers; row softmax over p via small LDS partials;
// emits similarity^T[q][p] = softmax_c + softmax_s directly as bf16.
__global__ void __launch_bounds__(256)
k_sigma_softmax(const uint16_t* __restrict__ segcT, const uint16_t* __restrict__ segc,
                const float* __restrict__ uvec, const float* __restrict__ vvec,
                const float* __restrict__ umm, const float* __restrict__ seg_ss,
                const float* __restrict__ edge_mm, uint16_t* __restrict__ simT) {
  __shared__ float red[16][129];
  __shared__ float rowDc[16], rowMy[16], rowDs[16];
  const int n  = blockIdx.y;
  const int q0 = blockIdx.x * 16;
  const int wv   = threadIdx.x >> 5;
  const int lane = threadIdx.x & 31;
  const int half = lane >> 4, col = lane & 15;
  const int pbase = wv * 128;
  const int g = wv * 16 + col;           // 0..127 partial slot per row

  const uint16_t* arow = segcT + ((size_t)n * HW + q0 + col) * Ss;
  const uint16_t* BtN  = segc + (size_t)n * Ss * HW;   // flat reinterpret [HW][64]

  v8f acc[8] = {};
  for (int k0 = 0; k0 < Ss; k0 += 32) {
    FragAB fa;
#pragma unroll
    for (int j = 0; j < 8; ++j) {
      const int ka = k0 + ((j & 3) << 1) + ((j >> 2) << 4) + (half << 3);
      fa.u[j] = *reinterpret_cast<const uint32_t*>(arow + ka);
    }
#pragma unroll
    for (int jt = 0; jt < 8; ++jt) {
      const uint16_t* brow = BtN + (size_t)(pbase + jt * 16 + col) * Ss;
      FragAB fb;
#pragma unroll
      for (int j = 0; j < 8; ++j) {
        const int kb = k0 + ((j & 3) << 1) + ((j >> 2) << 4) + (half << 3);
        fb.u[j] = *reinterpret_cast<const uint32_t*>(brow + kb);
      }
      acc[jt] = __builtin_amdgcn_wmma_f32_16x16x32_bf16(false, fa.v, false, fb.v,
                                                        (short)0, acc[jt], false, false);
    }
  }

  // per-lane auxiliaries
  float u8v[8], em8[8];
#pragma unroll
  for (int j = 0; j < 8; ++j) {
    const int p = pbase + j * 16 + col;
    u8v[j] = uvec[n * HW + p];
    em8[j] = edge_mm[n * HW + p];
  }
  const float umax = umm[2 * n], umin = umm[2 * n + 1];
  float vq8[8], sq8[8], mc8[8];
#pragma unroll
  for (int r = 0; r < 8; ++r) {
    const int q = q0 + r + 8 * half;
    const float vq = vvec[n * HW + q];
    vq8[r] = vq;
    sq8[r] = seg_ss[n * HW + q];
    mc8[r] = (vq >= 0.f) ? vq * umax : vq * umin;   // analytic column max (rank-1 branch)
  }

  // pass 1: channel-branch denominators + spatial-branch row maxima
  float dcp[8], myp[8];
#pragma unroll
  for (int r = 0; r < 8; ++r) {
    float dc = 0.f, my = -3.4e38f;
#pragma unroll
    for (int j = 0; j < 8; ++j) {
      dc += __expf(vq8[r] * u8v[j] - mc8[r]);
      my = fmaxf(my, em8[j] * sq8[r] * acc[j][r]);
    }
    dcp[r] = dc; myp[r] = my;
  }
#pragma unroll
  for (int r = 0; r < 8; ++r) red[r + 8 * half][g] = dcp[r];
  __syncthreads();
  if (threadIdx.x < 16) {
    float s = 0.f;
    for (int k = 0; k < 128; ++k) s += red[threadIdx.x][k];
    rowDc[threadIdx.x] = s;
  }
  __syncthreads();
#pragma unroll
  for (int r = 0; r < 8; ++r) red[r + 8 * half][g] = myp[r];
  __syncthreads();
  if (threadIdx.x < 16) {
    float s = -3.4e38f;
    for (int k = 0; k < 128; ++k) s = fmaxf(s, red[threadIdx.x][k]);
    rowMy[threadIdx.x] = s;
  }
  __syncthreads();

  // pass 2: spatial-branch exponentials (in place) + denominators
  float dsp[8];
#pragma unroll
  for (int r = 0; r < 8; ++r) {
    const float My = rowMy[r + 8 * half];
    float ds = 0.f;
#pragma unroll
    for (int j = 0; j < 8; ++j) {
      const float es = __expf(em8[j] * sq8[r] * acc[j][r] - My);
      acc[j][r] = es;
      ds += es;
    }
    dsp[r] = ds;
  }
#pragma unroll
  for (int r = 0; r < 8; ++r) red[r + 8 * half][g] = dsp[r];
  __syncthreads();
  if (threadIdx.x < 16) {
    float s = 0.f;
    for (int k = 0; k < 128; ++k) s += red[threadIdx.x][k];
    rowDs[threadIdx.x] = s;
  }
  __syncthreads();

  // pass 3: combine both softmaxes, write similarity^T bf16
#pragma unroll
  for (int r = 0; r < 8; ++r) {
    const int q = q0 + r + 8 * half;
    const float rDc = 1.f / rowDc[r + 8 * half];
    const float rDs = 1.f / rowDs[r + 8 * half];
    uint16_t* orow = simT + ((size_t)n * HW + q) * HW;
#pragma unroll
    for (int j = 0; j < 8; ++j) {
      const int p = pbase + j * 16 + col;
      const float ec = __expf(vq8[r] * u8v[j] - mc8[r]);
      orow[p] = f2bf(ec * rDc + acc[j][r] * rDs);
    }
  }
}

// seg_sim[c][q] = sum_p seg[c][p]*similarity[p][q]; store transposed bf16 [q][c]  (K=1024)
__global__ void __launch_bounds__(256)
k_gemm_sim(const uint16_t* __restrict__ seg_bf, const uint16_t* __restrict__ simT,
           uint16_t* __restrict__ segsimT) {
  const int n = blockIdx.y;
  const int tile = blockIdx.x * 8 + (threadIdx.x >> 5);
  const int m0 = (tile >> 6) * 16;   // c
  const int n0 = (tile & 63) * 16;   // q
  const uint16_t* A  = seg_bf + (size_t)n * Cc * HW;
  const uint16_t* Bt = simT   + (size_t)n * HW * HW;
  v8f acc = wmma_nt_tile(A, Bt, HW, HW, m0, n0, HW);
  const int lane = threadIdx.x & 31, half = lane >> 4, col = lane & 15;
#pragma unroll
  for (int r = 0; r < 8; ++r)
    segsimT[((size_t)n * HW + (n0 + col)) * Cc + (m0 + r + half*8)] = f2bf(acc[r]);
}

// out[o][l] = relu(W_g[o][:]·seg_sim[:][l] + b_g[o]) + 2*seg_ori[o][l]   (K=256)
__global__ void __launch_bounds__(256)
k_gemm_out(const uint16_t* __restrict__ Wg, const uint16_t* __restrict__ segsimT,
           const float* __restrict__ bg, const float* __restrict__ seg,
           float* __restrict__ out) {
  const int n = blockIdx.y;
  const int tile = blockIdx.x * 8 + (threadIdx.x >> 5);
  const int m0 = (tile >> 6) * 16;   // o
  const int n0 = (tile & 63) * 16;   // l
  const uint16_t* Bt = segsimT + (size_t)n * HW * Cc;
  v8f acc = wmma_nt_tile(Wg, Bt, Cc, Cc, m0, n0, Cc);
  const int lane = threadIdx.x & 31, half = lane >> 4, col = lane & 15;
#pragma unroll
  for (int r = 0; r < 8; ++r) {
    const int o = m0 + r + half * 8;
    const int l = n0 + col;
    const size_t gi = ((size_t)n * Cc + o) * HW + l;
    float v = acc[r] + bg[o];
    v = fmaxf(v, 0.f);
    out[gi] = v + 2.0f * seg[gi];
  }
}

// ---------------- host orchestration ----------------
extern "C" void kernel_launch(void* const* d_in, const int* in_sizes, int n_in,
                              void* d_out, int out_size, void* d_ws, size_t ws_size,
                              hipStream_t stream) {
  (void)in_sizes; (void)n_in; (void)out_size; (void)ws_size;
  const float* seg   = (const float*)d_in[0];
  const float* edge  = (const float*)d_in[1];
  const float* W_s1  = (const float*)d_in[2];
  const float* b_s1  = (const float*)d_in[3];
  const float* W_s11 = (const float*)d_in[4];
  const float* b_s11 = (const float*)d_in[5];
  const float* W_mlp = (const float*)d_in[6];
  const float* b_mlp = (const float*)d_in[7];
  const float* w_s2  = (const float*)d_in[8];
  const float* b_s2  = (const float*)d_in[9];
  const float* w_s3  = (const float*)d_in[10];
  const float* b_s3  = (const float*)d_in[11];
  const float* W_g   = (const float*)d_in[12];
  const float* b_g   = (const float*)d_in[13];
  float* out = (float*)d_out;

  char* w = (char*)d_ws;
  size_t off = 0;
  auto alloc = [&](size_t bytes) -> void* {
    void* p = (void*)(w + off);
    off += (bytes + 255) & ~(size_t)255;
    return p;
  };
  uint16_t* seg_bf   = (uint16_t*)alloc((size_t)Nn*Cc*HW*2);
  uint16_t* segT_bf  = (uint16_t*)alloc((size_t)Nn*HW*Cc*2);
  float*    segs_f   = (float*)   alloc((size_t)Nn*Ss*HW*4);
  uint16_t* segc_bf  = (uint16_t*)alloc((size_t)Nn*Ss*HW*2);
  uint16_t* segcT_bf = (uint16_t*)alloc((size_t)Nn*HW*Ss*2);
  uint16_t* Ws1_bf   = (uint16_t*)alloc((size_t)Ss*Cc*2);
  uint16_t* Ws11_bf  = (uint16_t*)alloc((size_t)Ss*Cc*2);
  uint16_t* Wg_bf    = (uint16_t*)alloc((size_t)Cc*Cc*2);
  float*    rowsum   = (float*)   alloc((size_t)Nn*Cc*4);
  float*    chatt    = (float*)   alloc((size_t)Nn*Ss*4);
  float*    seg_ss   = (float*)   alloc((size_t)Nn*HW*4);
  float*    edge_mm  = (float*)   alloc((size_t)Nn*HW*4);
  float*    uvec     = (float*)   alloc((size_t)Nn*HW*4);
  float*    vvec     = (float*)   alloc((size_t)Nn*HW*4);
  float*    umm      = (float*)   alloc((size_t)Nn*2*4);
  uint16_t* simT     = (uint16_t*)alloc((size_t)Nn*HW*HW*2);
  uint16_t* segsimT  = (uint16_t*)alloc((size_t)Nn*HW*Cc*2);

  // 1. zero row-sum accumulators (ws is not re-zeroed between calls)
  k_zero<<<dim3((Nn*Cc + 255)/256), 256, 0, stream>>>(rowsum, Nn*Cc);
  // 2. seg -> bf16 (+ transposed) + row sums
  k_trans<<<dim3(HW/32, Cc/32, Nn), dim3(32,8), 0, stream>>>(seg, seg_bf, segT_bf, rowsum);
  // 3. max-pools over channels
  k_prep_cols<<<dim3(HW/256, Nn), 256, 0, stream>>>(seg, edge, w_s2, b_s2, w_s3, b_s3,
                                                    seg_ss, edge_mm);
  // 4. weight conversions
  k_f2bf<<<dim3((Ss*Cc + 255)/256), 256, 0, stream>>>(W_s1,  Ws1_bf,  Ss*Cc);
  k_f2bf<<<dim3((Ss*Cc + 255)/256), 256, 0, stream>>>(W_s11, Ws11_bf, Ss*Cc);
  k_f2bf<<<dim3((Cc*Cc + 255)/256), 256, 0, stream>>>(W_g,   Wg_bf,   Cc*Cc);
  // 5. channel attention
  k_chatt<<<dim3(Nn), Ss, 0, stream>>>(rowsum, W_mlp, b_mlp, chatt);
  // 6/7. projection GEMMs (WMMA)
  k_gemm_sc<<<dim3(32, Nn), 256, 0, stream>>>(Ws1_bf,  segT_bf, b_s1,  segs_f, nullptr, nullptr, 0);
  k_gemm_sc<<<dim3(32, Nn), 256, 0, stream>>>(Ws11_bf, segT_bf, b_s11, nullptr, segc_bf, segcT_bf, 1);
  // 8. rank-1 channel-affinity vectors
  k_uv<<<dim3(Nn), 1024, 0, stream>>>(segs_f, chatt, uvec, vvec, umm);
  // 9. fused sigma GEMM (WMMA) + dual softmax -> similarity^T (bf16), no fp32 roundtrip
  k_sigma_softmax<<<dim3(HW/16, Nn), 256, 0, stream>>>(segcT_bf, segc_bf, uvec, vvec, umm,
                                                       seg_ss, edge_mm, simT);
  // 10. seg @ similarity GEMM (WMMA), transposed output
  k_gemm_sim<<<dim3(128, Nn), 256, 0, stream>>>(seg_bf, simT, segsimT);
  // 11. W_g GEMM (WMMA) + relu + residuals -> fp32 output
  k_gemm_out<<<dim3(128, Nn), 256, 0, stream>>>(Wg_bf, segsimT, b_g, seg, out);
}
